// BaselineDistiller_73126113181817
// MI455X (gfx1250) — matrix operand order repaired and checked by reading, not compiled, not run on hardware
//
#include <hip/hip_runtime.h>
#include <hip/hip_bf16.h>

typedef __attribute__((ext_vector_type(16))) _Float16 v16h;
typedef __attribute__((ext_vector_type(8)))  _Float16 v8h;
typedef __attribute__((ext_vector_type(8)))  float    v8f;

typedef __attribute__((ext_vector_type(4))) unsigned int u32x4;
typedef __attribute__((ext_vector_type(8))) int          i32x8;
typedef __attribute__((ext_vector_type(4))) int          i32x4;

constexpr int kB  = 16384;
constexpr int kNE = 32;
constexpr int kE  = 128;
constexpr int kH  = 256;

// LDS row pitches in f16 units (rows padded by 32B to spread banks)
constexpr int kP1 = kE + 16;   // 144 (W1^T rows: K along E) -> 288B pitch
constexpr int kP2 = kH + 16;   // 272 (W2^T rows: K along H) -> 544B pitch
constexpr int kPH = kH + 16;   // 272 (h rows)

constexpr int kWREG = kH * kP1 * 2;          // 73728 B (>= W2^T region 128*544=69632)
constexpr int kHREG = 8 * 16 * kPH * 2;      // 69632 B (8 waves * 16 rows)
constexpr int kREDO = kWREG + kHREG;         // reduction scratch offset
constexpr int kSMEM = kREDO + 1024;          // 144384 B dynamic LDS

// ---------------------------------------------------------------------------
// Convert weights to f16 and transpose so WMMA B-fragments are K-contiguous.
//   W1T[n][h][e] = W1[n][e][h];   W2T[n][e][h] = W2[n][h][e]
// ---------------------------------------------------------------------------
__global__ __launch_bounds__(256) void prep_weights(const float* __restrict__ W1,
                                                    const float* __restrict__ W2,
                                                    _Float16* __restrict__ W1T,
                                                    _Float16* __restrict__ W2T) {
  const int idx = blockIdx.x * 256 + threadIdx.x;   // [0, kNE*kH*kE)
  const int n   = idx >> 15;                        // kE*kH = 32768
  const int e1 = idx & (kE - 1);
  const int h1 = (idx >> 7) & (kH - 1);
  W1T[idx] = (_Float16)W1[((size_t)(n * kE + e1)) * kH + h1];
  const int h2 = idx & (kH - 1);
  const int e2 = (idx >> 8) & (kE - 1);
  W2T[idx] = (_Float16)W2[((size_t)(n * kH + h2)) * kE + e2];
}

// ---------------------------------------------------------------------------
// TDM: 2D tiled copy global -> LDS with row padding (D# per ISA 08 §8).
// dims/strides in DWORD units (data_size = 4B).
// pad_interval: pad after (2<<pi) DWORDs; pad_amount: (pa+1) DWORDs inserted.
// ---------------------------------------------------------------------------
#if __has_builtin(__builtin_amdgcn_tensor_load_to_lds)
#define HAVE_TDM 1
__device__ __forceinline__ void tdm_load_2d(unsigned lds_off, const void* gptr,
                                            unsigned d0_dw, unsigned d1_rows,
                                            unsigned pad_interval,
                                            unsigned pad_amount) {
  const unsigned long long ga = (unsigned long long)(size_t)gptr;
  u32x4 g0;
  g0[0] = 1u;                                           // count=1, user mode
  g0[1] = lds_off;                                      // D#.lds_addr (bytes)
  g0[2] = (unsigned)(ga & 0xFFFFFFFFu);                 // global_addr[31:0]
  g0[3] = ((unsigned)(ga >> 32) & 0x01FFFFFFu)          // global_addr[56:32]
          | 0x80000000u;                                // type = 2 ("image")
  i32x8 g1;
  g1[0] = (int)((2u << 16)                              // data_size = 4B
                | (1u << 20)                            // pad_enable
                | (pad_interval << 22)
                | (pad_amount << 25));
  g1[1] = (int)((d0_dw & 0xFFFFu) << 16);               // tensor_dim0 lo16
  g1[2] = (int)(((d0_dw >> 16) & 0xFFFFu)               // tensor_dim0 hi16
                | ((d1_rows & 0xFFFFu) << 16));         // tensor_dim1 lo16
  g1[3] = (int)(((d1_rows >> 16) & 0xFFFFu)             // tensor_dim1 hi16
                | (d0_dw << 16));                       // tile_dim0
  g1[4] = (int)(d1_rows & 0xFFFFu);                     // tile_dim1 (tile_dim2=0)
  g1[5] = (int)d0_dw;                                   // tensor_dim0_stride lo32
  g1[6] = 0;
  g1[7] = 0;
  const i32x4 gz = {0, 0, 0, 0};
#if defined(__clang_major__) && (__clang_major__ >= 23)
  const i32x8 gz8 = {0, 0, 0, 0, 0, 0, 0, 0};
  __builtin_amdgcn_tensor_load_to_lds(g0, g1, gz, gz, gz8, 0);
#else
  __builtin_amdgcn_tensor_load_to_lds(g0, g1, gz, gz, 0);
#endif
}
#else
#define HAVE_TDM 0
#endif

// ---------------------------------------------------------------------------
// Fused: GEMM1 -> bias -> exact GELU -> GEMM2 -> bias -> sq-err partial sums.
// One expert per blockIdx.y, 128 rows per block (8 waves x 16 rows).
// ---------------------------------------------------------------------------
__global__ __launch_bounds__(256, 1) void moe_distill_fused(
    const float* __restrict__ feat, const float* __restrict__ tgt,
    const float* __restrict__ b1g, const float* __restrict__ b2g,
    const _Float16* __restrict__ W1T, const _Float16* __restrict__ W2T,
    float* __restrict__ partials) {
  extern __shared__ char smem[];
  _Float16* wlds = (_Float16*)smem;             // weight staging region
  _Float16* hlds = (_Float16*)(smem + kWREG);   // per-wave h tiles
  float*    red  = (float*)(smem + kREDO);      // 256-float reduction

  const int n    = blockIdx.y;
  const int tid  = threadIdx.x;
  const int wave = tid >> 5;
  const int lane = tid & 31;
  const int hsel = lane >> 4;     // which 16-lane half
  const int l16  = lane & 15;
  const int r0   = blockIdx.x * 128 + wave * 16;

  // ---- stage W1T[n]: 256 rows x 256B, padded to 288B pitch ----
#if HAVE_TDM
  if (wave == 0) {
    // 64 DWORDs/row, 256 rows; pad 8 DW (32B) after every 64 DW (256B)
    tdm_load_2d((unsigned)(size_t)wlds, W1T + (size_t)n * kH * kE,
                /*d0_dw=*/64, /*rows=*/256, /*pad_interval=*/5, /*pad_amount=*/7);
    __builtin_amdgcn_s_wait_tensorcnt(0);
  }
#else
  {
    const uint4* src = (const uint4*)(W1T + (size_t)n * kH * kE);
    uint4* dst = (uint4*)wlds;                  // row pitch = 288/16 = 18 uint4
    #pragma unroll
    for (int it = 0; it < 16; ++it) {
      const int i = it * 256 + tid;
      const int r = i >> 4, j = i & 15;
      dst[r * 18 + j] = src[i];
    }
  }
#endif
  __syncthreads();

  // ---- GEMM1: h[16 x 256] = feat[16 x 128] * W1 ----
  v8f acc[16] = {};
  const float* frow = feat + ((size_t)(r0 + l16) * kNE + n) * kE;
  #pragma unroll
  for (int kk = 0; kk < 4; ++kk) {
    const int k0 = kk * 32;
    // A-layout (16-bit, 16x32): lane half h holds K = k0+16h..+15 (contiguous)
    const float4 f0 = *(const float4*)(frow + k0 + hsel * 16);
    const float4 f1 = *(const float4*)(frow + k0 + hsel * 16 + 4);
    const float4 f2 = *(const float4*)(frow + k0 + hsel * 16 + 8);
    const float4 f3 = *(const float4*)(frow + k0 + hsel * 16 + 12);
    v16h a;
    a[0]=(_Float16)f0.x;  a[1]=(_Float16)f0.y;  a[2]=(_Float16)f0.z;  a[3]=(_Float16)f0.w;
    a[4]=(_Float16)f1.x;  a[5]=(_Float16)f1.y;  a[6]=(_Float16)f1.z;  a[7]=(_Float16)f1.w;
    a[8]=(_Float16)f2.x;  a[9]=(_Float16)f2.y;  a[10]=(_Float16)f2.z; a[11]=(_Float16)f2.w;
    a[12]=(_Float16)f3.x; a[13]=(_Float16)f3.y; a[14]=(_Float16)f3.z; a[15]=(_Float16)f3.w;
    #pragma unroll
    for (int t = 0; t < 16; ++t) {
      // B-layout (16-bit, 32x16): lane = col N, 16 contiguous K at k0+16*hsel
      const v16h b = *(const v16h*)(wlds + (t * 16 + l16) * kP1 + k0 + hsel * 16);
      acc[t] = __builtin_amdgcn_wmma_f32_16x16x32_f16(false, a, false, b,
                                                      (short)0, acc[t], false, false);
    }
  }

  // ---- bias + exact GELU, stage h as f16 in this wave's LDS tile ----
  _Float16* hw = hlds + wave * (16 * kPH);
  #pragma unroll
  for (int t = 0; t < 16; ++t) {
    const int col = t * 16 + l16;
    const float bias = b1g[n * kH + col];
    #pragma unroll
    for (int v = 0; v < 8; ++v) {
      const float x = acc[t][v] + bias;
      const float g = 0.5f * x * (1.0f + erff(x * 0.70710678118654752f));
      hw[(v + hsel * 8) * kPH + col] = (_Float16)g;   // C-layout: M = v + 8*hsel
    }
  }

  __syncthreads();   // everyone done reading W1T

  // ---- stage W2T[n]: 128 rows x 512B, padded to 544B pitch ----
#if HAVE_TDM
  if (wave == 0) {
    // 128 DWORDs/row, 128 rows; pad 8 DW (32B) after every 128 DW (512B)
    tdm_load_2d((unsigned)(size_t)wlds, W2T + (size_t)n * kH * kE,
                /*d0_dw=*/128, /*rows=*/128, /*pad_interval=*/6, /*pad_amount=*/7);
    __builtin_amdgcn_s_wait_tensorcnt(0);
  }
#else
  {
    const uint4* src = (const uint4*)(W2T + (size_t)n * kH * kE);
    uint4* dst = (uint4*)wlds;                  // row pitch = 544/16 = 34 uint4
    #pragma unroll
    for (int it = 0; it < 16; ++it) {
      const int i = it * 256 + tid;
      const int r = i >> 5, j = i & 31;
      dst[r * 34 + j] = src[i];
    }
  }
#endif
  __syncthreads();

  // ---- GEMM2: pred[16 x 128] = gelu_h[16 x 256] * W2 ----
  v8f acc2[8] = {};
  const _Float16* ha = hw + l16 * kPH;
  #pragma unroll
  for (int kk = 0; kk < 8; ++kk) {
    const int k0 = kk * 32;
    const v8h lo = *(const v8h*)(ha + k0 + hsel * 16);
    const v8h hi = *(const v8h*)(ha + k0 + hsel * 16 + 8);
    const v16h a = __builtin_shufflevector(lo, hi, 0,1,2,3,4,5,6,7,
                                                   8,9,10,11,12,13,14,15);
    #pragma unroll
    for (int t = 0; t < 8; ++t) {
      const v16h b = *(const v16h*)(wlds + (t * 16 + l16) * kP2 + k0 + hsel * 16);
      acc2[t] = __builtin_amdgcn_wmma_f32_16x16x32_f16(false, a, false, b,
                                                       (short)0, acc2[t], false, false);
    }
  }

  // ---- bias2, squared error vs target ----
  float sum = 0.0f;
  #pragma unroll
  for (int t = 0; t < 8; ++t) {
    const int e = t * 16 + l16;
    const float bias = b2g[n * kE + e];
    #pragma unroll
    for (int v = 0; v < 8; ++v) {
      const size_t row = (size_t)(r0 + v + hsel * 8);
      const float tv = tgt[(row * kNE + n) * kE + e];
      const float d = acc2[t][v] + bias - tv;
      sum += d * d;
    }
  }

  red[tid] = sum;
  __syncthreads();
  #pragma unroll
  for (int s = 128; s > 0; s >>= 1) {
    if (tid < s) red[tid] += red[tid + s];
    __syncthreads();
  }
  if (tid == 0) partials[blockIdx.y * gridDim.x + blockIdx.x] = red[0];
}

// ---------------------------------------------------------------------------
// Deterministic final reduction over per-workgroup partials -> mean.
// ---------------------------------------------------------------------------
__global__ __launch_bounds__(256) void final_reduce(const float* __restrict__ partials,
                                                    float* __restrict__ out, int np) {
  __shared__ float s[256];
  float acc = 0.0f;
  for (int i = threadIdx.x; i < np; i += 256) acc += partials[i];
  s[threadIdx.x] = acc;
  __syncthreads();
  for (int st = 128; st > 0; st >>= 1) {
    if ((int)threadIdx.x < st) s[threadIdx.x] += s[threadIdx.x + st];
    __syncthreads();
  }
  if (threadIdx.x == 0)
    out[0] = s[0] * (1.0f / ((float)kB * (float)kNE * (float)kE));
}

extern "C" void kernel_launch(void* const* d_in, const int* in_sizes, int n_in,
                              void* d_out, int out_size, void* d_ws, size_t ws_size,
                              hipStream_t stream) {
  const float* feat = (const float*)d_in[0];   // [B, NE, E]
  const float* tgt  = (const float*)d_in[1];   // [B, NE, E]
  const float* W1   = (const float*)d_in[2];   // [NE, E, H]
  const float* b1   = (const float*)d_in[3];   // [NE, H]
  const float* W2   = (const float*)d_in[4];   // [NE, H, E]
  const float* b2   = (const float*)d_in[5];   // [NE, E]

  char* ws = (char*)d_ws;
  _Float16* W1T   = (_Float16*)(ws);                                // 2 MB
  _Float16* W2T   = (_Float16*)(ws + (size_t)kNE * kH * kE * 2);    // 2 MB
  float* partials = (float*)   (ws + (size_t)kNE * kH * kE * 4);    // 16 KB

  const int nElemW = kNE * kH * kE;            // 1,048,576
  prep_weights<<<nElemW / 256, 256, 0, stream>>>(W1, W2, W1T, W2T);

  dim3 grid(kB / 128, kNE);                    // 128 x 32 = 4096 workgroups
  moe_distill_fused<<<grid, 256, kSMEM, stream>>>(feat, tgt, b1, b2, W1T, W2T,
                                                  partials);

  final_reduce<<<1, 256, 0, stream>>>(partials, (float*)d_out,
                                      (kB / 128) * kNE);
}
